// TSClsSparseTransformerModel_41420664603323
// MI455X (gfx1250) — compile-verified, matrix operand order checked by reading, not run on hardware
//
#include <hip/hip_runtime.h>
#include <math.h>

typedef __attribute__((ext_vector_type(16))) __bf16 v16bf;
typedef __attribute__((ext_vector_type(8)))  __bf16 v8bf;
typedef __attribute__((ext_vector_type(8)))  float  v8f;

// ---------- fp32 -> bf16 (round-to-nearest-even) ----------
__device__ __forceinline__ __bf16 f2bf(float x) {
  unsigned u = __builtin_bit_cast(unsigned, x);
  u += 0x7FFFu + ((u >> 16) & 1u);
  unsigned short s = (unsigned short)(u >> 16);
  return __builtin_bit_cast(__bf16, s);
}

// Load one lane's slice of a 16x32 bf16 WMMA fragment from bf16 memory.
// CDNA5 16-bit A layout: lanes 0-15 hold K {0..7,16..23} of row=lane,
// lanes 16-31 hold K {8..15,24..31} of row=lane-16. rowp points at (row, k0+kh):
// two contiguous 16-byte chunks -> 2x global_load_b128, no conversion.
__device__ __forceinline__ v16bf load_frag(const __bf16* __restrict__ rowp) {
  v8bf lo = *(const v8bf*)(rowp);
  v8bf hi = *(const v8bf*)(rowp + 16);
  return __builtin_shufflevector(lo, hi, 0,1,2,3,4,5,6,7,8,9,10,11,12,13,14,15);
}

__device__ __forceinline__ void store_out(float* C, size_t idx, float v)  { C[idx] = v; }
__device__ __forceinline__ void store_out(__bf16* C, size_t idx, float v) { C[idx] = f2bf(v); }

// ---------- WMMA GEMM: C[M,N] = act(A[M,K](bf16) * W[N,K](bf16)^T + bias[N]) ----------
// Block = 256 threads = 8 waves (4m x 2n). Wave tile 32x64 (2 A-frags x 4 B-frags,
// 8 accumulators). Block tile 128x128. grid = (M/128, N/128). All dims are exact
// multiples -> no predication, EXEC all-ones for WMMA.
template <bool RELU, typename OutT>
__global__ __launch_bounds__(256) void gemm_bias(
    const __bf16* __restrict__ A, int lda,
    const __bf16* __restrict__ W, int ldw,
    const float* __restrict__ bias,
    OutT* __restrict__ C, int ldc, int K)
{
  const int wid  = threadIdx.x >> 5;
  const int lane = threadIdx.x & 31;
  const int m0 = blockIdx.x * 128 + (wid >> 1) * 32;
  const int n0 = blockIdx.y * 128 + (wid & 1) * 64;

  const int r  = lane & 15;
  const int kh = (lane >> 4) << 3;   // 0 or 8

  const __bf16* Arow0 = A + (size_t)(m0 + r)      * lda + kh;
  const __bf16* Arow1 = A + (size_t)(m0 + 16 + r) * lda + kh;
  const __bf16* Wrow0 = W + (size_t)(n0 + r)      * ldw + kh;
  const __bf16* Wrow1 = W + (size_t)(n0 + 16 + r) * ldw + kh;
  const __bf16* Wrow2 = W + (size_t)(n0 + 32 + r) * ldw + kh;
  const __bf16* Wrow3 = W + (size_t)(n0 + 48 + r) * ldw + kh;

  v8f acc[2][4];
#pragma unroll
  for (int i = 0; i < 2; ++i)
#pragma unroll
    for (int j = 0; j < 4; ++j) acc[i][j] = (v8f){};

  for (int k0 = 0; k0 < K; k0 += 32) {
    v16bf a0 = load_frag(Arow0 + k0);
    v16bf a1 = load_frag(Arow1 + k0);
    v16bf b0 = load_frag(Wrow0 + k0);
    v16bf b1 = load_frag(Wrow1 + k0);
    v16bf b2 = load_frag(Wrow2 + k0);
    v16bf b3 = load_frag(Wrow3 + k0);
    acc[0][0] = __builtin_amdgcn_wmma_f32_16x16x32_bf16(false, a0, false, b0, (short)0, acc[0][0], false, false);
    acc[0][1] = __builtin_amdgcn_wmma_f32_16x16x32_bf16(false, a0, false, b1, (short)0, acc[0][1], false, false);
    acc[0][2] = __builtin_amdgcn_wmma_f32_16x16x32_bf16(false, a0, false, b2, (short)0, acc[0][2], false, false);
    acc[0][3] = __builtin_amdgcn_wmma_f32_16x16x32_bf16(false, a0, false, b3, (short)0, acc[0][3], false, false);
    acc[1][0] = __builtin_amdgcn_wmma_f32_16x16x32_bf16(false, a1, false, b0, (short)0, acc[1][0], false, false);
    acc[1][1] = __builtin_amdgcn_wmma_f32_16x16x32_bf16(false, a1, false, b1, (short)0, acc[1][1], false, false);
    acc[1][2] = __builtin_amdgcn_wmma_f32_16x16x32_bf16(false, a1, false, b2, (short)0, acc[1][2], false, false);
    acc[1][3] = __builtin_amdgcn_wmma_f32_16x16x32_bf16(false, a1, false, b3, (short)0, acc[1][3], false, false);
  }

  // C/D layout: element i of v8f -> row base + i + (lane>>4)*8, col base + (lane&15)
  const int nlo = lane & 15;
  const int mo  = (lane >> 4) << 3;  // 0 or 8
#pragma unroll
  for (int j = 0; j < 4; ++j) {
    const int nc = n0 + j * 16 + nlo;
    const float bia = bias[nc];
#pragma unroll
    for (int mi = 0; mi < 2; ++mi) {
      const int mbase = m0 + mi * 16 + mo;
#pragma unroll
      for (int i = 0; i < 8; ++i) {
        float v = acc[mi][j][i] + bia;
        if (RELU) v = fmaxf(v, 0.f);
        store_out(C, (size_t)(mbase + i) * ldc + nc, v);
      }
    }
  }
}

// ---------- Convert fp32 -> bf16, 4 elems/thread (n multiple of 1024) ----------
__global__ __launch_bounds__(256) void cvt_bf16(
    const float* __restrict__ in, __bf16* __restrict__ out)
{
  const size_t i = ((size_t)blockIdx.x * 256 + threadIdx.x) * 4;
  float4 v = *(const float4*)(in + i);
  out[i + 0] = f2bf(v.x); out[i + 1] = f2bf(v.y);
  out[i + 2] = f2bf(v.z); out[i + 3] = f2bf(v.w);
}

// ---------- Embedding + sqrt(E) scale + sinusoidal PE; writes fp32 + bf16 ----------
__global__ __launch_bounds__(256) void embed_kernel(
    const float* __restrict__ src, const float* __restrict__ embW,
    const float* __restrict__ embB, float* __restrict__ x, __bf16* __restrict__ xb)
{
  const int row = blockIdx.x;        // b*512 + s
  const int e   = threadIdx.x;       // 0..255
  const int s   = row & 511;
  float acc = embB[e];
  const float* sp = src + (size_t)row * 9;
  const float* wp = embW + (size_t)e * 9;
#pragma unroll
  for (int d = 0; d < 9; ++d) acc += sp[d] * wp[d];
  const float i2  = (float)(e & ~1);
  const float div = __expf(-i2 * (9.210340372f / 256.f)); // ln(10000)/E
  const float ang = (float)s * div;
  const float pe  = (e & 1) ? __cosf(ang) : __sinf(ang);
  const float r = acc * 16.0f + pe;                       // sqrt(256) = 16
  x[(size_t)row * 256 + e]  = r;
  xb[(size_t)row * 256 + e] = f2bf(r);
}

// ---------- x = LayerNorm(x + y)*g + b ; also writes bf16 mirror ----------
__global__ __launch_bounds__(256) void resid_ln(
    float* __restrict__ x, const float* __restrict__ y, __bf16* __restrict__ xb,
    const float* __restrict__ g, const float* __restrict__ bb)
{
  __shared__ float red[256];
  const int t = threadIdx.x;
  const size_t idx = (size_t)blockIdx.x * 256 + t;
  const float v = x[idx] + y[idx];
  red[t] = v; __syncthreads();
  for (int off = 128; off > 0; off >>= 1) { if (t < off) red[t] += red[t + off]; __syncthreads(); }
  const float mean = red[0] * (1.0f / 256.0f);
  __syncthreads();
  const float d = v - mean;
  red[t] = d * d; __syncthreads();
  for (int off = 128; off > 0; off >>= 1) { if (t < off) red[t] += red[t + off]; __syncthreads(); }
  const float var = red[0] * (1.0f / 256.0f);
  const float r = d * rsqrtf(var + 1e-5f) * g[t] + bb[t];
  x[idx]  = r;
  xb[idx] = f2bf(r);
}

// ---------- Sparse attention: only the last query row is non-trivial ----------
// One block per batch, one wave per head. qkv is bf16 [B*S, 768]. Computes the
// last-row softmax over keys j<511 (j=511 underflows to exactly 0 in the
// reference), writes head-averaged attention row to d_out, and overwrites the
// V slice of the last row so the O-proj GEMM reads the V section directly.
__global__ __launch_bounds__(256) void attn_sparse(
    __bf16* __restrict__ qkv, float* __restrict__ attnL)
{
  __shared__ float q_sh[8][32];
  __shared__ float a_sh[8][512];
  const int b = blockIdx.x;
  const int t = threadIdx.x;
  const int h = t >> 5, lane = t & 31;

  __bf16* base = qkv + (size_t)b * 512 * 768;
  q_sh[h][lane] = (float)base[(size_t)511 * 768 + h * 32 + lane];
  __syncthreads();

  const __bf16* kcol = base + 256 + h * 32;
  float sc[16];
  float smax = -1e30f;
  for (int it = 0; it < 16; ++it) {
    const int j = lane + it * 32;
    const __bf16* krow = kcol + (size_t)j * 768;
    float d = 0.f;
#pragma unroll
    for (int e = 0; e < 32; ++e) d += q_sh[h][e] * (float)krow[e];
    d *= 0.1767766953f;  // 1/sqrt(32)
    sc[it] = d;
    if (j < 511) smax = fmaxf(smax, d);
  }
  for (int off = 16; off > 0; off >>= 1) smax = fmaxf(smax, __shfl_xor(smax, off, 32));
  float w[16], ssum = 0.f;
  for (int it = 0; it < 16; ++it) {
    const int j = lane + it * 32;
    const float e = (j < 511) ? __expf(sc[it] - smax) : 0.f;
    w[it] = e; ssum += e;
  }
  for (int off = 16; off > 0; off >>= 1) ssum += __shfl_xor(ssum, off, 32);
  const float inv = 1.f / ssum;
  for (int it = 0; it < 16; ++it) a_sh[h][lane + it * 32] = w[it] * inv;
  __syncthreads();

  const __bf16* vcol = base + 512 + h * 32;
  float o = 0.f;
  for (int j = 0; j < 511; ++j) o += a_sh[h][j] * (float)vcol[(size_t)j * 768 + lane];
  base[(size_t)511 * 768 + 512 + h * 32 + lane] = f2bf(o);

  for (int j = t; j < 512; j += 256) {
    float m = 0.f;
#pragma unroll
    for (int hh = 0; hh < 8; ++hh) m += a_sh[hh][j];
    attnL[(size_t)b * 512 * 512 + (size_t)511 * 512 + j] = m * 0.125f;
  }
}

// ---------- Fill attns[L,B,S,S]: exact one-hot rows (i<511), zeros for row 511 ----------
__global__ __launch_bounds__(256) void fill_attns(float* __restrict__ attns)
{
  const size_t idx = ((size_t)blockIdx.x * 256 + threadIdx.x) * 4;
  const int j0 = (int)(idx & 511);
  const int i  = (int)((idx >> 9) & 511);
  float4 v;
  v.x = (i == j0     && i < 511) ? 1.f : 0.f;
  v.y = (i == j0 + 1 && i < 511) ? 1.f : 0.f;
  v.z = (i == j0 + 2 && i < 511) ? 1.f : 0.f;
  v.w = (i == j0 + 3 && i < 511) ? 1.f : 0.f;
  *(float4*)(attns + idx) = v;
}

// ---------- Head: xs[b,e] = sum_s relu(x[b,s,e]) ----------
__global__ __launch_bounds__(256) void colsum_relu(
    const float* __restrict__ x, float* __restrict__ xs)
{
  const int b = blockIdx.x, e = threadIdx.x;
  float acc = 0.f;
  for (int s = 0; s < 512; ++s) acc += fmaxf(x[((size_t)b * 512 + s) * 256 + e], 0.f);
  xs[b * 256 + e] = acc;
}

// ---------- out[b,c] = log_softmax_c( (1/S) * sum_e xs[b,e]*decW[c,e] ) ----------
__global__ __launch_bounds__(32) void decode_ls(
    const float* __restrict__ xs, const float* __restrict__ decW, float* __restrict__ out)
{
  __shared__ float logits[10];
  const int b = blockIdx.x, t = threadIdx.x;
  if (t < 10) {
    float acc = 0.f;
#pragma unroll 4
    for (int e = 0; e < 256; ++e) acc += xs[b * 256 + e] * decW[t * 256 + e];
    logits[t] = acc * (1.0f / 512.0f);
  }
  __syncthreads();
  float m = -1e30f;
  for (int c = 0; c < 10; ++c) m = fmaxf(m, logits[c]);
  float s = 0.f;
  for (int c = 0; c < 10; ++c) s += __expf(logits[c] - m);
  const float lse = m + __logf(s);
  if (t < 10) out[b * 10 + t] = logits[t] - lse;
}

extern "C" void kernel_launch(void* const* d_in, const int* in_sizes, int n_in,
                              void* d_out, int out_size, void* d_ws, size_t ws_size,
                              hipStream_t stream) {
  const float* src  = (const float*)d_in[0];
  const float* embW = (const float*)d_in[1];
  const float* embB = (const float*)d_in[2];
  const float* Wqkv = (const float*)d_in[3];
  const float* bqkv = (const float*)d_in[4];
  const float* Wo   = (const float*)d_in[5];
  const float* bo   = (const float*)d_in[6];
  const float* ln1g = (const float*)d_in[7];
  const float* ln1b = (const float*)d_in[8];
  const float* ln2g = (const float*)d_in[9];
  const float* ln2b = (const float*)d_in[10];
  const float* W1   = (const float*)d_in[11];
  const float* b1   = (const float*)d_in[12];
  const float* W2   = (const float*)d_in[13];
  const float* b2   = (const float*)d_in[14];
  const float* decW = (const float*)d_in[15];

  float* out   = (float*)d_out;
  float* attns = out + 320;                        // [4,32,512,512] after [32,10]

  // workspace layout
  float* xbuf = (float*)d_ws;                      // 16384*256 fp32
  float* buf2 = xbuf + (size_t)16384 * 256;        // 16384*256 fp32
  float* xs   = buf2 + (size_t)16384 * 256;        // 32*256 fp32
  __bf16* xbufb = (__bf16*)(xs + 8192);            // 16384*256 bf16
  __bf16* buf1b = xbufb + (size_t)16384 * 256;     // 16384*1024 bf16 (qkv / ffn hidden)
  __bf16* Wqkvb = buf1b + (size_t)16384 * 1024;    // 4*768*256
  __bf16* Wob   = Wqkvb + (size_t)4 * 768 * 256;   // 4*256*256
  __bf16* W1b   = Wob   + (size_t)4 * 256 * 256;   // 4*1024*256
  __bf16* W2b   = W1b   + (size_t)4 * 1024 * 256;  // 4*256*1024

  // one-time weight conversion to bf16 (element counts are multiples of 1024)
  cvt_bf16<<<(4 * 768 * 256) / 1024, 256, 0, stream>>>(Wqkv, Wqkvb);
  cvt_bf16<<<(4 * 256 * 256) / 1024, 256, 0, stream>>>(Wo, Wob);
  cvt_bf16<<<(4 * 1024 * 256) / 1024, 256, 0, stream>>>(W1, W1b);
  cvt_bf16<<<(4 * 256 * 1024) / 1024, 256, 0, stream>>>(W2, W2b);

  fill_attns<<<32768, 256, 0, stream>>>(attns);
  embed_kernel<<<16384, 256, 0, stream>>>(src, embW, embB, xbuf, xbufb);

  for (int l = 0; l < 4; ++l) {
    // qkv = x @ Wqkv^T + bqkv  -> bf16 [16384, 768]
    gemm_bias<false, __bf16><<<dim3(128, 6), 256, 0, stream>>>(
        xbufb, 256, Wqkvb + (size_t)l * 768 * 256, 256, bqkv + l * 768, buf1b, 768, 256);
    // sparse attention (patches last-row V in place, writes attn rows)
    attn_sparse<<<32, 256, 0, stream>>>(buf1b, attns + (size_t)l * 32 * 512 * 512);
    // o @ Wo^T + bo : A = V section of qkv (lda=768) -> fp32 for residual
    gemm_bias<false, float><<<dim3(128, 2), 256, 0, stream>>>(
        buf1b + 512, 768, Wob + (size_t)l * 256 * 256, 256, bo + l * 256, buf2, 256, 256);
    resid_ln<<<16384, 256, 0, stream>>>(xbuf, buf2, xbufb, ln1g + l * 256, ln1b + l * 256);
    // FFN
    gemm_bias<true, __bf16><<<dim3(128, 8), 256, 0, stream>>>(
        xbufb, 256, W1b + (size_t)l * 1024 * 256, 256, b1 + l * 1024, buf1b, 1024, 256);
    gemm_bias<false, float><<<dim3(128, 2), 256, 0, stream>>>(
        buf1b, 1024, W2b + (size_t)l * 256 * 1024, 1024, b2 + l * 256, buf2, 256, 1024);
    resid_ln<<<16384, 256, 0, stream>>>(xbuf, buf2, xbufb, ln2g + l * 256, ln2b + l * 256);
  }

  colsum_relu<<<32, 256, 0, stream>>>(xbuf, xs);
  decode_ls<<<32, 32, 0, stream>>>(xs, decW, out);
}
